// SFA_5600637354583
// MI455X (gfx1250) — matrix operand order repaired and checked by reading, not compiled, not run on hardware
//
#include <hip/hip_runtime.h>
#include <hip/hip_bf16.h>

typedef _Float16 v16h __attribute__((ext_vector_type(16)));
typedef _Float16 v8h  __attribute__((ext_vector_type(8)));
typedef float    v8f  __attribute__((ext_vector_type(8)));

#define BATCH 4
#define HH    180
#define WW    180
#define HWSZ  (HH * WW)          // 32400
#define PH    182                // spatially padded (zero halo) for branch-free 3x3 taps
#define PW    182
#define PHW   (PH * PW)          // 33124
#define NBORD 724                // 2*182 + 2*180 border pixels per image
#define NC    10
#define CCAT  320                // 306 padded to multiple of 32
#define CHID  64
#define CAL   256
#define CF1   128

// ---------------- workspace layout (bytes, all sizes multiples of 64) ----------------
#define OFF_NA   ((size_t)0)                          // 4*32400*10*4   = 5,184,000
#define OFF_NB   ((size_t)5184000)                    // 5,184,000
#define OFF_CAT  ((size_t)10368000)                   // 4*33124*320*2  = 84,797,440
#define OFF_AL   OFF_CAT                              // 4*33124*256*2  = 67,837,952 (aliases CAT)
#define OFF_HID  ((size_t)95165440)                   // 4*33124*64*2   = 16,959,488
#define OFF_F1   ((size_t)112124928)                  // 4*33124*128*2  = 33,918,976
#define OFF_WP1  ((size_t)146043904)                  // 4*90*512*2     = 368,640
#define OFF_WP3  ((size_t)146412544)                  // 8*72*512*2     = 589,824
#define OFF_WP4  ((size_t)147002368)                  // 8*36*512*2     = 294,912
// total ~147.3 MB

__device__ __forceinline__ float sigm(float v) { return 1.0f / (1.0f + expf(-v)); }

// padded NHWC pixel base (elements) for flattened batch-pixel q
__device__ __forceinline__ size_t pix_base(int q) {
    int b = q / HWSZ, p = q % HWSZ;
    int y = p / WW,  x = p % WW;
    return (size_t)b * PHW + (size_t)(y + 1) * PW + (x + 1);
}

// ---- 0. zero the spatial halo of a padded NHWC f16 buffer ----
__global__ __launch_bounds__(256) void zero_border_kernel(_Float16* __restrict__ buf, int C)
{
    size_t gid = (size_t)blockIdx.x * blockDim.x + threadIdx.x;
    if (gid >= (size_t)BATCH * NBORD * C) return;
    int c  = (int)(gid % C);
    int r  = (int)(gid / C);
    int bi = r % NBORD;
    int b  = r / NBORD;
    int row, col;
    if (bi < PW)            { row = 0;      col = bi; }
    else if (bi < 2 * PW)   { row = PH - 1; col = bi - PW; }
    else { int t = bi - 2 * PW; row = 1 + t / 2; col = (t & 1) ? (PW - 1) : 0; }
    buf[((size_t)b * PHW + (size_t)row * PW + col) * C + c] = (_Float16)0.0f;
}

// ---- 1. sigmoid(transposed heatmaps) + per-pixel L2 norm over 10 channels ----
// heatmaps are (B, NC, W, H); transposed value[b,c,h,w] = hm[((b*NC+c)*W + w)*H + h]
__global__ __launch_bounds__(256) void norm_heat_kernel(
    const float* __restrict__ hp, const float* __restrict__ hi,
    float* __restrict__ nA, float* __restrict__ nB)
{
    int gid = blockIdx.x * blockDim.x + threadIdx.x;
    if (gid >= BATCH * HWSZ) return;
    int b = gid / HWSZ, p = gid % HWSZ;
    int y = p / WW, x = p % WW;
    float a[NC], bb[NC], sa = 0.f, sb = 0.f;
    for (int c = 0; c < NC; ++c) {
        size_t idx = ((size_t)(b * NC + c) * WW + x) * HH + y;
        float va = sigm(hp[idx]); a[c]  = va; sa += va * va;
        float vb = sigm(hi[idx]); bb[c] = vb; sb += vb * vb;
    }
    sa = fmaxf(sqrtf(sa), 1e-12f);
    sb = fmaxf(sqrtf(sb), 1e-12f);
    float* da = nA + (size_t)gid * NC;
    float* db = nB + (size_t)gid * NC;
    for (int c = 0; c < NC; ++c) { da[c] = a[c] / sa; db[c] = bb[c] / sb; }
}

// ---- 2. fill concat (padded NHWC f16): [0..127]=pts, [153..280]=img, [306..319]=0 ----
__global__ __launch_bounds__(256) void concat_base_kernel(
    const float* __restrict__ pts, const float* __restrict__ img,
    _Float16* __restrict__ cat)
{
    size_t gid = (size_t)blockIdx.x * blockDim.x + threadIdx.x;
    if (gid >= (size_t)BATCH * CCAT * HWSZ) return;
    int p = (int)(gid % HWSZ);
    size_t r = gid / HWSZ;
    int c = (int)(r % CCAT);
    int b = (int)(r / CCAT);
    _Float16* d = cat + pix_base(b * HWSZ + p) * CCAT + c;
    if (c < 128)                   *d = (_Float16)pts[((size_t)(b * 128 + c)) * HWSZ + p];
    else if (c >= 153 && c < 281)  *d = (_Float16)img[((size_t)(b * 128 + (c - 153))) * HWSZ + p];
    else if (c >= 306)             *d = (_Float16)0.0f;
    // 128..152 and 281..305 written by costvol_kernel
}

// ---- 3. dual 5x5 cost volumes into concat channels [128..152] and [281..305] ----
__global__ __launch_bounds__(256) void costvol_kernel(
    const float* __restrict__ nA, const float* __restrict__ nB,
    _Float16* __restrict__ cat)
{
    int gid = blockIdx.x * blockDim.x + threadIdx.x;
    if (gid >= BATCH * HWSZ) return;
    int b = gid / HWSZ, p = gid % HWSZ;
    int y = p / WW, x = p % WW;
    float a[NC], bb[NC];
    const float* ac = nA + (size_t)gid * NC;
    const float* bc = nB + (size_t)gid * NC;
    for (int c = 0; c < NC; ++c) { a[c] = ac[c]; bb[c] = bc[c]; }
    _Float16* dst = cat + pix_base(gid) * CCAT;
    for (int k = 0; k < 25; ++k) {
        int dy = k / 5 - 2, dx = k % 5 - 2;
        int yy = y + dy, xx = x + dx;
        float cp = 0.f, ci = 0.f;
        if ((unsigned)yy < (unsigned)HH && (unsigned)xx < (unsigned)WW) {
            size_t np = (size_t)b * HWSZ + (size_t)yy * WW + xx;
            const float* an = nA + np * NC;
            const float* bn = nB + np * NC;
            for (int c = 0; c < NC; ++c) { cp += a[c] * bn[c]; ci += bb[c] * an[c]; }
        }
        dst[128 + k] = (_Float16)cp;
        dst[281 + k] = (_Float16)ci;
    }
}

// ---- 4. pack OIHW f32 weights into WMMA 16x16x32 f16 A-fragment order ----
// layout: [cout_tile][kpos*(Cpad/32) chunks][32 lanes][16 halves]
// A-matrix per-lane k order used by this code (contiguous; B loads use the same):
//   lane<16 : halves 0..15 -> K 0..15
//   lane>=16: halves 0..15 -> K 16..31
__global__ __launch_bounds__(256) void pack_wmma_a_kernel(
    const float* __restrict__ w, _Float16* __restrict__ wp,
    int Cout, int CinReal, int Cpad)
{
    int nkc = 9 * (Cpad >> 5);
    size_t total = (size_t)(Cout >> 4) * nkc * 512;
    size_t gid = (size_t)blockIdx.x * blockDim.x + threadIdx.x;
    if (gid >= total) return;
    int h    = (int)(gid & 15);
    int lane = (int)((gid >> 4) & 31);
    size_t rest = gid >> 9;
    int kcT  = (int)(rest % nkc);
    int tile = (int)(rest / nkc);
    int ml = lane & 15, hf = lane >> 4;
    int K = kcT * 32 + 16 * hf + h;
    int kpos = K / Cpad, c = K % Cpad;
    int ky = kpos / 3, kx = kpos % 3;
    int co = tile * 16 + ml;
    float v = (c < CinReal) ? w[(((size_t)co * CinReal + c) * 3 + ky) * 3 + kx] : 0.f;
    wp[gid] = (_Float16)v;
}

// ---- 5. implicit-GEMM 3x3 conv, padded NHWC f16 in, 2x2 register tile ----
// each wave: 2 cout tiles x 2 pixel tiles (32 ch x 32 px), 4 accumulators,
// each A/B fragment reused twice -> 2 b128 loads per v_wmma, zero branches.
template <int CIN, int COUT, bool OUT32>
__global__ __launch_bounds__(256) void conv3x3_wmma2x2_kernel(
    const _Float16* __restrict__ in, const _Float16* __restrict__ wp,
    const float* __restrict__ bias, void* __restrict__ outv, int relu)
{
    constexpr int NKC   = CIN / 32;       // k-chunks per tap
    constexpr int NCOT2 = COUT / 32;      // cout tile-pairs
    const int totalWaves = (BATCH * HWSZ / 32) * NCOT2;
    const int wid  = (int)((blockIdx.x * blockDim.x + threadIdx.x) >> 5);
    if (wid >= totalWaves) return;        // wave-uniform exit, EXEC stays all-1s
    const int lane = threadIdx.x & 31;
    const int ct   = wid % NCOT2;         // channels ct*32 .. ct*32+31
    const int wt   = wid / NCOT2;         // 32-pixel tile over flattened B*HWSZ
    const int n    = lane & 15;
    const int hf   = lane >> 4;
    const int q0   = wt * 32 + n;
    const int q1   = q0 + 16;

    const _Float16* in0 = in + pix_base(q0) * CIN + hf * 16;
    const _Float16* in1 = in + pix_base(q1) * CIN + hf * 16;
    const _Float16* wb0 = wp + (size_t)(2 * ct)     * 9 * NKC * 512 + lane * 16;
    const _Float16* wb1 = wp + (size_t)(2 * ct + 1) * 9 * NKC * 512 + lane * 16;

    v8f acc00 = {}, acc01 = {}, acc10 = {}, acc11 = {};
    #pragma unroll
    for (int kpos = 0; kpos < 9; ++kpos) {
        const int off = ((kpos / 3 - 1) * PW + (kpos % 3 - 1)) * CIN;  // compile-time per kpos
        #pragma unroll
        for (int kc = 0; kc < NKC; ++kc) {
            v16h a0 = *(const v16h*)(wb0 + (kpos * NKC + kc) * 512);
            v16h a1 = *(const v16h*)(wb1 + (kpos * NKC + kc) * 512);
            v16h b0 = *(const v16h*)(in0 + off + kc * 32);
            v16h b1 = *(const v16h*)(in1 + off + kc * 32);
            acc00 = __builtin_amdgcn_wmma_f32_16x16x32_f16(false, a0, false, b0, (short)0, acc00, false, false);
            acc01 = __builtin_amdgcn_wmma_f32_16x16x32_f16(false, a0, false, b1, (short)0, acc01, false, false);
            acc10 = __builtin_amdgcn_wmma_f32_16x16x32_f16(false, a1, false, b0, (short)0, acc10, false, false);
            acc11 = __builtin_amdgcn_wmma_f32_16x16x32_f16(false, a1, false, b1, (short)0, acc11, false, false);
        }
    }

    // D layout: vgpr r -> channel (tileBase + hf*8 + r), pixel = lane%16 of the tile
    v8f accs[4] = { acc00, acc01, acc10, acc11 };
    #pragma unroll
    for (int t = 0; t < 2; ++t) {
        const int co = ct * 32 + t * 16 + hf * 8;
        #pragma unroll
        for (int pi = 0; pi < 2; ++pi) {
            const int q = pi ? q1 : q0;
            v8f ac = accs[t * 2 + pi];
            if constexpr (OUT32) {
                float* out = (float*)outv;            // unpadded f32 NCHW (final layer)
                const int b = q / HWSZ, p = q % HWSZ;
                #pragma unroll
                for (int r = 0; r < 8; ++r)
                    out[((size_t)(b * COUT + co + r)) * HWSZ + p] = ac[r] + bias[co + r];
            } else {
                _Float16* out = (_Float16*)outv;      // padded NHWC f16
                v8h o;
                #pragma unroll
                for (int r = 0; r < 8; ++r) {
                    float v = ac[r] + bias[co + r];
                    if (relu) v = fmaxf(v, 0.0f);
                    o[r] = (_Float16)v;
                }
                *(v8h*)(out + pix_base(q) * COUT + co) = o;
            }
        }
    }
}

// ---- 6. fused: conv3x3 64->4 (VALU) + tanh*5 flow + bilinear border sampling ----
// hidden is padded NHWC f16 (zero halo) -> no bounds checks in the 3x3 loop.
// aligned (padded NHWC f16): channels [0..127]=aligned_img, [128..255]=aligned_pts
__device__ __forceinline__ void sample_border_128(
    const float* __restrict__ im, int b, float sx, float sy,
    _Float16* __restrict__ dst)
{
    sx = fminf(fmaxf(sx, 0.f), (float)(WW - 1));
    sy = fminf(fmaxf(sy, 0.f), (float)(HH - 1));
    float x0f = floorf(sx), y0f = floorf(sy);
    int x0 = (int)x0f, y0 = (int)y0f;
    int x1 = min(x0 + 1, WW - 1), y1 = min(y0 + 1, HH - 1);
    float wx = sx - x0f, wy = sy - y0f;
    float w00 = (1.f - wx) * (1.f - wy), w01 = wx * (1.f - wy);
    float w10 = (1.f - wx) * wy,         w11 = wx * wy;
    const float* base = im + (size_t)b * 128 * HWSZ;
    size_t i00 = (size_t)y0 * WW + x0, i01 = (size_t)y0 * WW + x1;
    size_t i10 = (size_t)y1 * WW + x0, i11 = (size_t)y1 * WW + x1;
    for (int c = 0; c < 128; ++c) {
        const float* pc = base + (size_t)c * HWSZ;
        dst[c] = (_Float16)(pc[i00] * w00 + pc[i01] * w01 + pc[i10] * w10 + pc[i11] * w11);
    }
}

__global__ __launch_bounds__(256) void flow_sample_kernel(
    const _Float16* __restrict__ hid, const float* __restrict__ w2,
    const float* __restrict__ b2, const float* __restrict__ pts,
    const float* __restrict__ img, _Float16* __restrict__ aligned)
{
    __shared__ float w2s[4 * 64 * 9];
    for (int i = threadIdx.x; i < 4 * 64 * 9; i += blockDim.x) w2s[i] = w2[i];
    __syncthreads();

    int gid = blockIdx.x * blockDim.x + threadIdx.x;
    if (gid >= BATCH * HWSZ) return;
    int b = gid / HWSZ, p = gid % HWSZ;
    int y = p / WW, x = p % WW;

    float acc[4] = { b2[0], b2[1], b2[2], b2[3] };
    const _Float16* hb = hid + ((size_t)b * PHW + (size_t)y * PW + x) * CHID; // padded (y,x)->tap (y+ky,x+kx)
    #pragma unroll
    for (int kpos = 0; kpos < 9; ++kpos) {
        int ky = kpos / 3, kx = kpos % 3;
        const _Float16* hp = hb + (size_t)(ky * PW + kx) * CHID;
        for (int c = 0; c < CHID; ++c) {
            float hv = (float)hp[c];
            acc[0] += hv * w2s[((0 * 64 + c) * 3 + ky) * 3 + kx];
            acc[1] += hv * w2s[((1 * 64 + c) * 3 + ky) * 3 + kx];
            acc[2] += hv * w2s[((2 * 64 + c) * 3 + ky) * 3 + kx];
            acc[3] += hv * w2s[((3 * 64 + c) * 3 + ky) * 3 + kx];
        }
    }
    float f0 = tanhf(acc[0]) * 5.0f;  // pts dx
    float f1 = tanhf(acc[1]) * 5.0f;  // pts dy
    float f2 = tanhf(acc[2]) * 5.0f;  // img dx
    float f3 = tanhf(acc[3]) * 5.0f;  // img dy

    _Float16* dst = aligned + pix_base(gid) * CAL;
    sample_border_128(img, b, (float)x + f2, (float)y + f3, dst);        // aligned_i first
    sample_border_128(pts, b, (float)x + f0, (float)y + f1, dst + 128);  // then aligned_p
}

extern "C" void kernel_launch(void* const* d_in, const int* in_sizes, int n_in,
                              void* d_out, int out_size, void* d_ws, size_t ws_size,
                              hipStream_t stream) {
    (void)in_sizes; (void)n_in; (void)out_size; (void)ws_size;
    const float* pts = (const float*)d_in[0];
    const float* img = (const float*)d_in[1];
    const float* hmp = (const float*)d_in[2];
    const float* hmi = (const float*)d_in[3];
    const float* fw1 = (const float*)d_in[4];
    const float* fb1 = (const float*)d_in[5];
    const float* fw2 = (const float*)d_in[6];
    const float* fb2 = (const float*)d_in[7];
    const float* gw1 = (const float*)d_in[8];
    const float* gb1 = (const float*)d_in[9];
    const float* gw2 = (const float*)d_in[10];
    const float* gb2 = (const float*)d_in[11];
    float* out = (float*)d_out;

    char* ws = (char*)d_ws;
    float*    nA      = (float*)(ws + OFF_NA);
    float*    nB      = (float*)(ws + OFF_NB);
    _Float16* cat     = (_Float16*)(ws + OFF_CAT);
    _Float16* aligned = (_Float16*)(ws + OFF_AL);   // aliases cat (cat dead after conv1)
    _Float16* hid     = (_Float16*)(ws + OFF_HID);
    _Float16* f1buf   = (_Float16*)(ws + OFF_F1);
    _Float16* wp1     = (_Float16*)(ws + OFF_WP1);
    _Float16* wp3     = (_Float16*)(ws + OFF_WP3);
    _Float16* wp4     = (_Float16*)(ws + OFF_WP4);

    const dim3 blk(256);
    const int pix = BATCH * HWSZ;                            // 129,600
    auto gb = [](size_t total) { return (unsigned)((total + 255) / 256); };

    // zero halos (cat + hid before conv1; f1 before conv3/4 reads)
    zero_border_kernel<<<gb((size_t)BATCH * NBORD * CCAT), blk, 0, stream>>>(cat, CCAT);
    zero_border_kernel<<<gb((size_t)BATCH * NBORD * CHID), blk, 0, stream>>>(hid, CHID);
    zero_border_kernel<<<gb((size_t)BATCH * NBORD * CF1),  blk, 0, stream>>>(f1buf, CF1);

    // heatmap sigmoid + L2 normalize
    norm_heat_kernel<<<gb(pix), blk, 0, stream>>>(hmp, hmi, nA, nB);

    // build concat (pts / img channels + zero pad channels)
    concat_base_kernel<<<gb((size_t)BATCH * CCAT * HWSZ), blk, 0, stream>>>(pts, img, cat);

    // cost volumes into concat
    costvol_kernel<<<gb(pix), blk, 0, stream>>>(nA, nB, cat);

    // weight packing for WMMA A-fragments
    pack_wmma_a_kernel<<<gb(4 * 90 * 512), blk, 0, stream>>>(fw1, wp1, 64, 306, CCAT);
    pack_wmma_a_kernel<<<gb(8 * 72 * 512), blk, 0, stream>>>(gw1, wp3, 128, 256, CAL);
    pack_wmma_a_kernel<<<gb(8 * 36 * 512), blk, 0, stream>>>(gw2, wp4, 128, 128, CF1);

    // conv1: concat(320) -> hidden(64), ReLU.  waves = 4050*2 = 8100
    conv3x3_wmma2x2_kernel<CCAT, CHID, false>
        <<<gb(8100 * 32), blk, 0, stream>>>(cat, wp1, fb1, hid, 1);

    // aligned halo: zero after conv1 consumed cat (aligned aliases cat)
    zero_border_kernel<<<gb((size_t)BATCH * NBORD * CAL), blk, 0, stream>>>(aligned, CAL);

    // conv2 + tanh flow + bilinear sampling -> aligned(256)
    flow_sample_kernel<<<gb(pix), blk, 0, stream>>>(hid, fw2, fb2, pts, img, aligned);

    // conv3: aligned(256) -> fused1(128), ReLU.  waves = 4050*4 = 16200
    conv3x3_wmma2x2_kernel<CAL, CF1, false>
        <<<gb(16200 * 32), blk, 0, stream>>>(aligned, wp3, gb1, f1buf, 1);

    // conv4: fused1(128) -> out f32 NCHW.  waves = 16200
    conv3x3_wmma2x2_kernel<CF1, CF1, true>
        <<<gb(16200 * 32), blk, 0, stream>>>(f1buf, wp4, gb2, out, 0);
}
